// GNNLinkPredictor_82772609728846
// MI455X (gfx1250) — matrix operand order repaired and checked by reading, not compiled, not run on hardware
//
#include <hip/hip_runtime.h>
#include <hip/hip_bf16.h>
#include <math.h>

// ---------------------------------------------------------------------------
// CDNA5 (gfx1250) wave32 WMMA types
// ---------------------------------------------------------------------------
typedef __bf16 bf16_t;
typedef __attribute__((ext_vector_type(16))) __bf16 v16bf;
typedef __attribute__((ext_vector_type(8)))  float  v8f;

#define WAVE 32
#define GTILES 8   // row tiles per block, GCN gemm
#define PTILES 8   // pair tiles per block, edge MLP

// Convert float4 -> 4 packed bf16 in LDS
__device__ __forceinline__ void st_bf4(bf16_t* dst, float4 v) {
    dst[0] = (bf16_t)v.x; dst[1] = (bf16_t)v.y;
    dst[2] = (bf16_t)v.z; dst[3] = (bf16_t)v.w;
}

// K-swizzle: within each 32-wide K chunk, reorder 8-element sub-blocks
// [0-7][8-15][16-23][24-31] -> [0-7][16-23][8-15][24-31]  (2-bit bit-reverse)
// so a lane's A fragment (ISA 7.12.2 16-bit A layout) is 16 contiguous bf16.
__device__ __forceinline__ int kswz(int k) {
    int r = k & 31, s = r >> 3;
    int slot = ((s & 1) << 1) | (s >> 1);
    return (k & ~31) | (slot << 3) | (r & 7);
}

// A fragment from swizzled LDS tile: one aligned 32B vector load per fragment.
__device__ __forceinline__ v16bf load_a_frag_swz(const bf16_t* A, int ld, int kbase, int lane) {
    int m = lane & 15;
    int khalf = lane >> 4;
    return *(const v16bf*)(A + m * ld + kbase + khalf * 16);
}

// B fragment: 32x16 bf16 tile (K x N), row-major LDS [K][ld].
// lane 0-15 : col N=lane, K=0..7,16..23 ; lane16-31 : col N=lane-16, K=8..15,24..31
// (scalar assembly; hoisted outside the tile loop so it is paid once/block)
__device__ __forceinline__ v16bf load_b_frag(const bf16_t* B, int ld, int kbase, int coloff, int lane) {
    int n = lane & 15;
    int khalf = lane >> 4;
    v16bf b;
#pragma unroll
    for (int v = 0; v < 8; ++v) {
        int k = kbase + khalf * 8 + v * 2 + ((v >> 2) << 3);
        b[2 * v]     = B[k * ld + coloff + n];
        b[2 * v + 1] = B[(k + 1) * ld + coloff + n];
    }
    return b;
}

// ---------------------------------------------------------------------------
// degree / normalization
// ---------------------------------------------------------------------------
__global__ void deg_init_kernel(float* deg, int N) {
    int i = blockIdx.x * blockDim.x + threadIdx.x;
    if (i < N) deg[i] = 1.0f;               // self-loop
}

__global__ void deg_count_kernel(const int* __restrict__ edst, float* deg, int E) {
    int e = blockIdx.x * blockDim.x + threadIdx.x;
    if (e < E) atomicAdd(&deg[edst[e]], 1.0f);
}

__global__ void deg_rsqrt_kernel(float* deg, int N) {
    int i = blockIdx.x * blockDim.x + threadIdx.x;
    if (i < N) deg[i] = rsqrtf(deg[i]);
}

__global__ void zero_kernel(float4* p, int n4) {
    int i = blockIdx.x * blockDim.x + threadIdx.x;
    if (i < n4) p[i] = make_float4(0.f, 0.f, 0.f, 0.f);
}

// ---------------------------------------------------------------------------
// GEMM: Out[N,128] = A[N,128] @ W[128,128], bf16 WMMA, f32 accumulate.
// Block = 256 threads (8 waves). W staged in LDS once, B fragments hoisted to
// registers, then GTILES 16-row strips per block with double-buffered A tile
// (one barrier per tile). Wave w owns cols [16w,16w+16).
// ---------------------------------------------------------------------------
__global__ void __launch_bounds__(256)
gemm128_kernel(const float* __restrict__ A, const float* __restrict__ W,
               float* __restrict__ Out, int Nrows) {
    __shared__ __align__(32) bf16_t As[2][16 * 128];   // 2 x 4 KB (K-swizzled)
    __shared__ bf16_t Bs[128 * 128];                   // 32 KB

    const int tid  = threadIdx.x;
    const int lane = tid & (WAVE - 1);
    const int wv   = tid >> 5;
    const int coloff = wv * 16;

    // stage W, 4096 float4 -> 16 per thread
#pragma unroll
    for (int it = 0; it < 16; ++it) {
        int idx = tid + it * 256;
        int r = idx >> 5, q = idx & 31;
        float4 v = ((const float4*)(W + (size_t)r * 128))[q];
        st_bf4(&Bs[r * 128 + q * 4], v);
    }
    __syncthreads();

    // hoist all B fragments into registers (paid once per block)
    v16bf bfrag[4];
#pragma unroll
    for (int kk = 0; kk < 4; ++kk)
        bfrag[kk] = load_b_frag(Bs, 128, kk * 32, coloff, lane);

    // per-lane staging / store coordinates
    const int sr = tid >> 5, sq = tid & 31;       // staging row / float4-col
    const int scol = kswz(sq * 4);
    const int n  = lane & 15;
    const int mb = (lane >> 4) * 8;

    const int tbase = blockIdx.x * GTILES;
    for (int t = 0; t < GTILES; ++t) {
        const int row0 = (tbase + t) * 16;
        if (row0 >= Nrows) break;                    // block-uniform
        bf16_t* Ab = As[t & 1];
        // stage A strip (swizzled), 512 float4 -> 2 per thread
#pragma unroll
        for (int it = 0; it < 2; ++it) {
            int r = sr + it * 8;
            int gr = row0 + r; if (gr >= Nrows) gr = Nrows - 1;
            float4 v = ((const float4*)(A + (size_t)gr * 128))[sq];
            st_bf4(&Ab[r * 128 + scol], v);
        }
        __syncthreads();

        v8f acc = {};
#pragma unroll
        for (int kk = 0; kk < 4; ++kk) {
            v16bf a = load_a_frag_swz(Ab, 128, kk * 32, lane);
            acc = __builtin_amdgcn_wmma_f32_16x16x32_bf16(false, a, false, bfrag[kk],
                                                          (short)0, acc, false, false);
        }

        // C/D layout: lane n = lane&15, M = v + 8*(lane>>4)
        if (row0 + 16 <= Nrows) {                    // uniform fast path
            float* op = Out + (size_t)(row0 + mb) * 128 + coloff + n;
#pragma unroll
            for (int v = 0; v < 8; ++v) op[v * 128] = acc[v];
        } else {
#pragma unroll
            for (int v = 0; v < 8; ++v) {
                int gr = row0 + mb + v;
                if (gr < Nrows) Out[(size_t)gr * 128 + coloff + n] = acc[v];
            }
        }
        // no trailing barrier: next tile writes the other A buffer; by the
        // time buffer t&1 is re-staged (t+2), everyone passed the t+1 barrier.
    }
}

// ---------------------------------------------------------------------------
// edge scatter: agg[dst] += xw[src] * dinv[src]*dinv[dst]  (one wave per edge)
// ---------------------------------------------------------------------------
__global__ void __launch_bounds__(256)
scatter_kernel(const float* __restrict__ xw, const float* __restrict__ dinv,
               const int* __restrict__ esrc, const int* __restrict__ edst,
               float* __restrict__ agg, int E) {
    int wv = threadIdx.x >> 5, lane = threadIdx.x & 31;
    int e = blockIdx.x * 8 + wv;
    if (e >= E) return;
    int s = esrc[e], d = edst[e];
    float nrm = dinv[s] * dinv[d];
    float4 v = ((const float4*)(xw + (size_t)s * 128))[lane];
    float* ad = agg + (size_t)d * 128 + lane * 4;
    atomicAdd(ad + 0, v.x * nrm);
    atomicAdd(ad + 1, v.y * nrm);
    atomicAdd(ad + 2, v.z * nrm);
    atomicAdd(ad + 3, v.w * nrm);
}

// h = maybe_relu(agg + xw * dinv^2 + b)
__global__ void finalize_kernel(const float4* __restrict__ agg, const float4* __restrict__ xw,
                                const float* __restrict__ dinv, const float* __restrict__ b,
                                float4* __restrict__ h, int N, int do_relu) {
    int idx = blockIdx.x * blockDim.x + threadIdx.x;   // one float4 (N*32 total)
    if (idx >= N * 32) return;
    int nrow = idx >> 5, q = idx & 31;
    float di = dinv[nrow];
    float sw = di * di;
    float4 a = agg[idx], x = xw[idx];
    const float4 bb = ((const float4*)b)[q];
    float4 r;
    r.x = a.x + x.x * sw + bb.x;
    r.y = a.y + x.y * sw + bb.y;
    r.z = a.z + x.z * sw + bb.z;
    r.w = a.w + x.w * sw + bb.w;
    if (do_relu) {
        r.x = fmaxf(r.x, 0.f); r.y = fmaxf(r.y, 0.f);
        r.z = fmaxf(r.z, 0.f); r.w = fmaxf(r.w, 0.f);
    }
    h[idx] = r;
}

// ---------------------------------------------------------------------------
// Edge MLP: feat=[h2[src]|h2[dst]] (16 x 256) -> relu(@Wp1[256,64]+bp1)
//           -> sigmoid(@Wp2[64]+bp2).
// Block = 128 threads (4 waves). Wp1 staged + B fragments hoisted once,
// then PTILES tiles of 16 pairs each.
// ---------------------------------------------------------------------------
__global__ void __launch_bounds__(128)
edge_mlp_kernel(const float* __restrict__ h2,
                const int* __restrict__ psrc, const int* __restrict__ pdst,
                const float* __restrict__ Wp1, const float* __restrict__ bp1,
                const float* __restrict__ Wp2, const float* __restrict__ bp2,
                float* __restrict__ out, int P) {
    __shared__ __align__(32) bf16_t Fs[16 * 256];   // 8 KB (K-swizzled features)
    __shared__ bf16_t Ws[256 * 64];                 // 32 KB Wp1
    __shared__ float  Zs[16 * 64];                  // 4 KB relu(z)
    __shared__ float  w2s[64];
    __shared__ float  b2s;

    const int tid  = threadIdx.x;
    const int lane = tid & 31;
    const int wv   = tid >> 5;
    const int coloff = wv * 16;

    // stage Wp1: 4096 float4 -> 32 per thread
#pragma unroll
    for (int it = 0; it < 32; ++it) {
        int idx = tid + it * 128;
        int r = idx >> 4, q = idx & 15;             // 16 float4 per 64-wide row
        float4 v = ((const float4*)(Wp1 + (size_t)r * 64))[q];
        st_bf4(&Ws[r * 64 + q * 4], v);
    }
    if (tid < 16) ((float4*)w2s)[tid] = ((const float4*)Wp2)[tid];
    if (tid == 0) b2s = bp2[0];
    __syncthreads();

    // hoist B fragments (8 x v16bf = 64 VGPRs, paid once per block)
    v16bf bfrag[8];
#pragma unroll
    for (int kk = 0; kk < 8; ++kk)
        bfrag[kk] = load_b_frag(Ws, 64, kk * 32, coloff, lane);

    const float bias1 = bp1[coloff + (lane & 15)];

    // per-lane staging coordinates: 64 float4 per 256-wide row, 2 rows/pass
    const int fr = tid >> 6, fq = tid & 63;
    const int fcol = fq * 4;
    const int fswz = kswz(fcol);
    const int n  = lane & 15;
    const int mb = (lane >> 4) * 8;

    const int tbase = blockIdx.x * PTILES;
    for (int t = 0; t < PTILES; ++t) {
        const int p0 = (tbase + t) * 16;
        if (p0 >= P) break;                         // block-uniform
        // stage gathered features (swizzled): 1024 float4 -> 8 per thread
#pragma unroll
        for (int it = 0; it < 8; ++it) {
            int r = fr + it * 2;
            int p = p0 + r; if (p >= P) p = P - 1;
            const float* src = (fcol < 128)
                ? (h2 + (size_t)psrc[p] * 128 + fcol)
                : (h2 + (size_t)pdst[p] * 128 + (fcol - 128));
            float4 v = *(const float4*)src;
            st_bf4(&Fs[r * 256 + fswz], v);
        }
        __syncthreads();

        v8f acc = {};
#pragma unroll
        for (int kk = 0; kk < 8; ++kk) {
            v16bf a = load_a_frag_swz(Fs, 256, kk * 32, lane);
            acc = __builtin_amdgcn_wmma_f32_16x16x32_bf16(false, a, false, bfrag[kk],
                                                          (short)0, acc, false, false);
        }
#pragma unroll
        for (int v = 0; v < 8; ++v)
            Zs[(mb + v) * 64 + coloff + n] = fmaxf(acc[v] + bias1, 0.f);
        __syncthreads();   // Zs ready; also covers Fs WAR for next tile

        if (tid < 16) {
            int p = p0 + tid;
            if (p < P) {
                const float4* zr = (const float4*)(Zs + tid * 64);
                const float4* wr = (const float4*)w2s;
                float dot = b2s;
#pragma unroll
                for (int j = 0; j < 16; ++j) {
                    float4 z = zr[j], w = wr[j];
                    dot += z.x * w.x + z.y * w.y + z.z * w.z + z.w * w.w;
                }
                out[p] = 1.0f / (1.0f + __expf(-dot));
            }
        }
    }
}

// ---------------------------------------------------------------------------
// launch
// ---------------------------------------------------------------------------
extern "C" void kernel_launch(void* const* d_in, const int* in_sizes, int n_in,
                              void* d_out, int out_size, void* d_ws, size_t ws_size,
                              hipStream_t stream) {
    const float* x    = (const float*)d_in[0];
    const int*   ei   = (const int*)  d_in[1];
    const int*   ep   = (const int*)  d_in[2];
    const float* W1   = (const float*)d_in[3];
    const float* b1   = (const float*)d_in[4];
    const float* W2   = (const float*)d_in[5];
    const float* b2   = (const float*)d_in[6];
    const float* Wp1  = (const float*)d_in[7];
    const float* bp1  = (const float*)d_in[8];
    const float* Wp2  = (const float*)d_in[9];
    const float* bp2  = (const float*)d_in[10];
    float* out = (float*)d_out;

    const int N = in_sizes[0] / 128;
    const int E = in_sizes[1] / 2;
    const int P = in_sizes[2] / 2;

    const int* esrc = ei;
    const int* edst = ei + E;
    const int* psrc = ep;
    const int* pdst = ep + P;

    // workspace layout (floats)
    float* ws = (float*)d_ws;
    size_t off = 0;
    float* dinv = ws + off; off += (size_t)((N + 15) & ~15);
    float* xw   = ws + off; off += (size_t)N * 128;
    float* agg  = ws + off; off += (size_t)N * 128;
    float* h1   = ws + off; off += (size_t)N * 128;
    float* h2   = ws + off; off += (size_t)N * 128;
    (void)ws_size; (void)n_in; (void)out_size;

    const int NH4 = N * 32;               // float4 count of an [N,128] buffer
    const int ggrid = (N + 16 * GTILES - 1) / (16 * GTILES);
    const int pgrid = (P + 16 * PTILES - 1) / (16 * PTILES);

    // normalization
    deg_init_kernel <<<(N + 255) / 256, 256, 0, stream>>>(dinv, N);
    deg_count_kernel<<<(E + 255) / 256, 256, 0, stream>>>(edst, dinv, E);
    deg_rsqrt_kernel<<<(N + 255) / 256, 256, 0, stream>>>(dinv, N);

    // ---- layer 1 ----
    gemm128_kernel<<<ggrid, 256, 0, stream>>>(x, W1, xw, N);
    zero_kernel<<<(NH4 + 255) / 256, 256, 0, stream>>>((float4*)agg, NH4);
    scatter_kernel<<<(E + 7) / 8, 256, 0, stream>>>(xw, dinv, esrc, edst, agg, E);
    finalize_kernel<<<(NH4 + 255) / 256, 256, 0, stream>>>(
        (const float4*)agg, (const float4*)xw, dinv, b1, (float4*)h1, N, 1);

    // ---- layer 2 ----
    gemm128_kernel<<<ggrid, 256, 0, stream>>>(h1, W2, xw, N);
    zero_kernel<<<(NH4 + 255) / 256, 256, 0, stream>>>((float4*)agg, NH4);
    scatter_kernel<<<(E + 7) / 8, 256, 0, stream>>>(xw, dinv, esrc, edst, agg, E);
    finalize_kernel<<<(NH4 + 255) / 256, 256, 0, stream>>>(
        (const float4*)agg, (const float4*)xw, dinv, b2, (float4*)h2, N, 0);

    // ---- edge MLP ----
    edge_mlp_kernel<<<pgrid, 128, 0, stream>>>(
        h2, psrc, pdst, Wp1, bp1, Wp2, bp2, out, P);
}
